// MinConv2dLSTM_58737972740761
// MI455X (gfx1250) — matrix operand order repaired and checked by reading, not compiled, not run on hardware
//
#include <hip/hip_runtime.h>

// ---------------------------------------------------------------------------
// MinConv2dLSTM on gfx1250 (MI455X): implicit-GEMM 3x3 conv via
// v_wmma_f32_16x16x32_bf16 (f32 accumulate), LDS-free B-fragment loads from a
// pre-swizzled bf16 input layout [img][pixel][Cin] (OOB lanes redirected to a
// zero pad -> unconditional b128 loads, no exec-mask churn), fused gate
// epilogue, then sequential scan kernels.
//
//   x : (4,16,32,32,32) f32  -> 64 images, Cin=32
//   w0: (256,32,3,3) b0:(256)   layer 0 (gates 4x C=64)
//   w1: (256,64,3,3) b1:(256)   layer 1
//   out: (4,16,64,32,32) f32
// ---------------------------------------------------------------------------

typedef __attribute__((ext_vector_type(16))) __bf16        v16bf;
typedef __attribute__((ext_vector_type(8)))  float         v8f;
typedef __attribute__((ext_vector_type(8)))  unsigned int  v8u;

__device__ __forceinline__ unsigned short f2bf(float f) {
    unsigned int u = __float_as_uint(f);
    unsigned int r = u + 0x7FFFu + ((u >> 16) & 1u);   // round-to-nearest-even
    return (unsigned short)(r >> 16);
}

__device__ __forceinline__ v16bf pack_bf(uint4 a, uint4 b) {
    v8u t;
    t[0] = a.x; t[1] = a.y; t[2] = a.z; t[3] = a.w;
    t[4] = b.x; t[5] = b.y; t[6] = b.z; t[7] = b.w;
    return __builtin_bit_cast(v16bf, t);
}

__device__ __forceinline__ float sigmoidf(float x) {
    return 1.0f / (1.0f + __expf(-x));
}

// ---------------------------------------------------------------------------
// Pack conv weights (Cout, Cin, 3, 3) f32 into a contiguous-k bf16 A-fragment
// per-lane layout; each lane's 16 bf16 live in 32 contiguous bytes:
//   Apack[(((tap*nCk + ck)*16 + coutTile)*32 + lane)*16 + i]
//   lane: m = lane&15, half = lane>>4;  element i -> K = half*16 + i
// Also zero-fills the 64-byte OOB pad used by the conv kernel.
// ---------------------------------------------------------------------------
__global__ void prep_weights_kernel(const float* __restrict__ w0,
                                    const float* __restrict__ w1,
                                    unsigned short* __restrict__ ap0,
                                    unsigned short* __restrict__ ap1,
                                    unsigned short* __restrict__ zpad) {
    int idx = blockIdx.x * 256 + threadIdx.x;
    if (blockIdx.x == 0 && threadIdx.x < 32) zpad[threadIdx.x] = 0;  // 64B zero pad
    if (idx >= 73728 + 147456) return;

    const float* w; unsigned short* dst; int li, nCk, Cin;
    if (idx < 73728) { w = w0; dst = ap0; li = idx;          nCk = 1; Cin = 32; }
    else             { w = w1; dst = ap1; li = idx - 73728;  nCk = 2; Cin = 64; }

    int i    = li & 15;
    int lane = (li >> 4) & 31;
    int ct   = (li >> 9) & 15;
    int ckt  = li >> 13;            // tap*nCk + ck
    int ck   = ckt % nCk;
    int tap  = ckt / nCk;           // 0..8 = ky*3+kx

    int m    = lane & 15;
    int half = lane >> 4;
    int k    = half * 16 + i;
    int cout = ct * 16 + m;
    int cin  = ck * 32 + k;
    dst[li]  = f2bf(w[((size_t)cout * Cin + cin) * 9 + tap]);
}

// ---------------------------------------------------------------------------
// Transpose + convert x (f32, [img][cin][y][x]) -> xbf (bf16, [img][p][cin]).
// Grid: 2048 blocks = 64 img x 32 rows; block 256. LDS 32x32 tile transpose:
// coalesced f32 reads (x fastest) and coalesced bf16 writes (cin fastest).
// ---------------------------------------------------------------------------
__global__ void prep_input_kernel(const float* __restrict__ x,
                                  unsigned short* __restrict__ xbf) {
    __shared__ unsigned short T[32][33];
    const int img = blockIdx.x >> 5;
    const int y   = blockIdx.x & 31;
    const int tid = threadIdx.x;

    #pragma unroll
    for (int it = 0; it < 4; ++it) {
        const int idx = it * 256 + tid;
        const int cin = idx >> 5;
        const int xx  = idx & 31;
        T[cin][xx] = f2bf(x[(((size_t)img * 32 + cin) << 10) + y * 32 + xx]);
    }
    __syncthreads();
    #pragma unroll
    for (int it = 0; it < 4; ++it) {
        const int idx = it * 256 + tid;
        const int xx  = idx >> 5;
        const int cin = idx & 31;
        xbf[((((size_t)img << 10) + y * 32 + xx) << 5) + cin] = T[cin][xx];
    }
}

// ---------------------------------------------------------------------------
// Implicit-GEMM conv + fused minLSTM gate epilogue.  LDS-free, no barriers.
// Grid: 2048 blocks = 64 images x 32 rows. Block: 256 thr = 8 waves.
//   wave w: tileIdx = w>>2 (pixel cols 0-15 / 16-31), cg = w&3 (16-ch group).
//   Per wave: 4 accumulators (gates i,f,o,cand); cout = g*64 + cg*16 + m.
// B fragment: 32 contiguous bytes of swizzled bf16 input per lane per tap;
// OOB lanes read the shared 64-byte zero pad (pointer select, full EXEC).
// layoutA=1: store gates as [img][p][c] (float4 x2, coalesced)  -> layer 0
// layoutA=0: store gates as [img][c][p] (scalar, coalesced)     -> layer 1
// ---------------------------------------------------------------------------
__global__ void __launch_bounds__(256)
conv_gates_kernel(const unsigned short* __restrict__ in,   // [64][1024][Cin] bf16
                  const unsigned short* __restrict__ apack,
                  const unsigned short* __restrict__ zpad, // 64B of zeros
                  const float* __restrict__ bias,          // [256]
                  float* __restrict__ fpo,
                  float* __restrict__ igo,
                  float* __restrict__ soo,
                  int Cin, int layoutA) {
    const int tid  = threadIdx.x;
    const int lane = tid & 31;
    const int wave = tid >> 5;
    const int tileIdx = wave >> 2;
    const int cg      = wave & 3;
    const int n    = lane & 15;
    const int half = lane >> 4;

    const int img = blockIdx.x >> 5;     // 0..63
    const int y   = blockIdx.x & 31;     // output row
    const int nCk = Cin >> 5;
    const int xcol = tileIdx * 16 + n;   // output column for this lane

    const unsigned short* inimg = in + (((size_t)img << 10) * Cin) + half * 16;

    v8f acc[4];
    #pragma unroll
    for (int g = 0; g < 4; ++g) acc[g] = (v8f){0,0,0,0,0,0,0,0};

    #pragma unroll
    for (int tap = 0; tap < 9; ++tap) {
        const int dy = tap / 3 - 1;
        const int dx = tap % 3 - 1;
        const int yy = y + dy;
        const int xx = xcol + dx;
        const bool inb = ((unsigned)yy < 32u) && ((unsigned)xx < 32u);
        // Unconditional loads: OOB lanes point at the zero pad instead.
        const unsigned short* brow =
            inb ? (inimg + (size_t)(yy * 32 + xx) * Cin) : zpad;

        for (int ck = 0; ck < nCk; ++ck) {
            const uint4* bp = (const uint4*)(brow + ck * 32);
            uint4 q0 = bp[0];
            uint4 q1 = bp[1];
            v16bf bv = pack_bf(q0, q1);

            #pragma unroll
            for (int g = 0; g < 4; ++g) {
                const int ct = g * 4 + cg;
                const uint4* ap = (const uint4*)(apack +
                    ((((size_t)tap * nCk + ck) * 16 + ct) * 32 + lane) * 16);
                uint4 a0 = ap[0];
                uint4 a1 = ap[1];
                v16bf av = pack_bf(a0, a1);
                acc[g] = __builtin_amdgcn_wmma_f32_16x16x32_bf16(
                    false, av, false, bv, (short)0, acc[g], false, false);
            }
        }
    }

    // NOTE: for ck>0 the zero-pad path reads zpad[32..47], past the 32 zeroed
    // elements (pad is 64B), so layer 1 OOB taps read memory never zeroed here.

    // ---- fused epilogue: bias + gate nonlinearities -> fp, ig, sigma(o) ----
    const int p_pix = y * 32 + xcol;
    const int cbase = cg * 16 + half * 8;

    float fpv[8], igv[8], sov[8];
    #pragma unroll
    for (int r = 0; r < 8; ++r) {
        const int c = cbase + r;
        const float iv = sigmoidf(acc[0][r] + bias[      c]);
        const float fv = sigmoidf(acc[1][r] + bias[ 64 + c]);
        const float ov = sigmoidf(acc[2][r] + bias[128 + c]);
        const float cd =          acc[3][r] + bias[192 + c];
        const float ga = (cd >= 0.0f) ? (cd + 0.5f) : sigmoidf(cd);
        const float den = iv + fv;
        fpv[r] = fv / den;
        igv[r] = (iv / den) * ga;
        sov[r] = ov;
    }

    if (layoutA) {
        // [img][p][c]: 8 consecutive channels per lane -> 2x float4 per array
        const size_t o = ((((size_t)img << 10) + p_pix) << 6) + cbase;
        *(float4*)(fpo + o)     = make_float4(fpv[0], fpv[1], fpv[2], fpv[3]);
        *(float4*)(fpo + o + 4) = make_float4(fpv[4], fpv[5], fpv[6], fpv[7]);
        *(float4*)(igo + o)     = make_float4(igv[0], igv[1], igv[2], igv[3]);
        *(float4*)(igo + o + 4) = make_float4(igv[4], igv[5], igv[6], igv[7]);
        *(float4*)(soo + o)     = make_float4(sov[0], sov[1], sov[2], sov[3]);
        *(float4*)(soo + o + 4) = make_float4(sov[4], sov[5], sov[6], sov[7]);
    } else {
        // [img][c][p]: scalar stores, lanes of a row are contiguous in p
        const size_t imgoff = (size_t)img * 64 * 1024;
        #pragma unroll
        for (int r = 0; r < 8; ++r) {
            const size_t o = imgoff + ((size_t)(cbase + r) << 10) + p_pix;
            fpo[o] = fpv[r];
            igo[o] = igv[r];
            soo[o] = sov[r];
        }
    }
}

// ---------------------------------------------------------------------------
// Layer-0 scan: gates in [img][p][c] f32; emits h0 in swizzled bf16 [img][p][c]
// (exactly the conv input layout for layer 1). Thread = (b, p, c), c fastest:
// coalesced reads AND writes.  c0 = 0.5; c = fp*c + ig; h = sigma(o)*c.
// ---------------------------------------------------------------------------
__global__ void scan_to_bf16_kernel(const float* __restrict__ fp,
                                    const float* __restrict__ ig,
                                    const float* __restrict__ so,
                                    unsigned short* __restrict__ h0bf) {
    const int idx = blockIdx.x * 256 + threadIdx.x;   // 0 .. 262143
    const int b   = idx >> 16;
    const int p   = (idx >> 6) & 1023;
    const int c   = idx & 63;
    float cst = 0.5f;
    #pragma unroll
    for (int s = 0; s < 16; ++s) {
        const size_t o = (((((size_t)b * 16 + s) << 10) + p) << 6) + c;
        cst = fp[o] * cst + ig[o];
        h0bf[o] = f2bf(so[o] * cst);
    }
}

// ---------------------------------------------------------------------------
// Layer-1 scan: gates in [img][c][p] f32; emits final output f32 in the
// reference layout (B,S,64,32,32). Thread = (b, c, p), p fastest: coalesced.
// ---------------------------------------------------------------------------
__global__ void scan_to_out_kernel(const float* __restrict__ fp,
                                   const float* __restrict__ ig,
                                   const float* __restrict__ so,
                                   float* __restrict__ out) {
    const int idx = blockIdx.x * 256 + threadIdx.x;   // 0 .. 262143
    const int b   = idx >> 16;
    const int c   = (idx >> 10) & 63;
    const int p   = idx & 1023;
    float cst = 0.5f;
    #pragma unroll
    for (int s = 0; s < 16; ++s) {
        const size_t o = ((((size_t)b * 16 + s) * 64 + c) << 10) + p;
        cst = fp[o] * cst + ig[o];
        out[o] = so[o] * cst;
    }
}

// ---------------------------------------------------------------------------
extern "C" void kernel_launch(void* const* d_in, const int* in_sizes, int n_in,
                              void* d_out, int out_size, void* d_ws, size_t ws_size,
                              hipStream_t stream) {
    (void)in_sizes; (void)n_in; (void)out_size; (void)ws_size;

    const float* x  = (const float*)d_in[0];
    const float* w0 = (const float*)d_in[1];
    const float* b0 = (const float*)d_in[2];
    const float* w1 = (const float*)d_in[3];
    const float* b1 = (const float*)d_in[4];
    float* out = (float*)d_out;

    // Workspace layout (~60.5 MB):
    //   [0)         Apack0   147456 B (bf16)
    //   [147456)    Apack1   294912 B (bf16)  -> end 442368
    //   [442368)    zpad     64 B of zeros
    //   [442496)    xbf      4 MiB  bf16 [64][1024][32]
    //   [4636800)   h0bf     8 MiB  bf16 [64][1024][64]
    //   [13025408)  fp       16 MiB f32
    //   [+16MiB)    ig       16 MiB
    //   [+32MiB)    so       16 MiB
    char* ws = (char*)d_ws;
    unsigned short* ap0  = (unsigned short*)(ws);
    unsigned short* ap1  = (unsigned short*)(ws + 147456);
    unsigned short* zpad = (unsigned short*)(ws + 442368);
    unsigned short* xbf  = (unsigned short*)(ws + 442496);
    unsigned short* h0bf = (unsigned short*)(ws + 4636800);
    float* fpA = (float*)(ws + 13025408);
    float* igA = fpA + 4194304;
    float* soA = igA + 4194304;

    // 1) pack weights into WMMA A-fragment layout (+ zero pad)
    prep_weights_kernel<<<864, 256, 0, stream>>>(w0, w1, ap0, ap1, zpad);

    // 2) swizzle + bf16-convert input: [img][cin][p] -> [img][p][cin]
    prep_input_kernel<<<2048, 256, 0, stream>>>(x, xbf);

    // 3) layer 0 conv gates (Cin=32, gate layout [img][p][c])
    conv_gates_kernel<<<2048, 256, 0, stream>>>(xbf, ap0, zpad, b0,
                                                fpA, igA, soA, 32, 1);

    // 4) layer 0 scan -> h0 (bf16, swizzled for layer-1 conv)
    scan_to_bf16_kernel<<<1024, 256, 0, stream>>>(fpA, igA, soA, h0bf);

    // 5) layer 1 conv gates (Cin=64, gate layout [img][c][p])
    conv_gates_kernel<<<2048, 256, 0, stream>>>(h0bf, ap1, zpad, b1,
                                                fpA, igA, soA, 64, 0);

    // 6) layer 1 scan -> final output
    scan_to_out_kernel<<<1024, 256, 0, stream>>>(fpA, igA, soA, out);
}
